// GIN_11252814315555
// MI455X (gfx1250) — matrix operand order repaired and checked by reading, not compiled, not run on hardware
//
#include <hip/hip_runtime.h>
#include <hip/hip_bf16.h>

// ---- CDNA5 WMMA vector types -------------------------------------------------
typedef __attribute__((ext_vector_type(16))) __bf16 v16bf;
typedef __attribute__((ext_vector_type(8)))  __bf16 v8bf;
typedef __attribute__((ext_vector_type(8)))  float  v8f;
typedef __attribute__((ext_vector_type(4)))  float  v4f;

union BFrag { v16bf v; v8bf h[2]; };

#define LN_EPS 1e-5f

// ds_swizzle_b32, group-of-32 XOR mode: lane' = ((lane & 0x1f) | 0) ^ xor.
// XOR of 1/2/4/8 stays inside each 16-lane half — exactly the lanes that hold
// one row's 16 columns in the WMMA C/D layout.  Mask must be a compile-time
// constant, so the butterfly step is a template parameter.
template <int M>
__device__ __forceinline__ void swz_step(float (&sv)[8], float (&qv)[8]) {
#pragma unroll
  for (int r = 0; r < 8; ++r)
    sv[r] += __int_as_float(
        __builtin_amdgcn_ds_swizzle(__float_as_int(sv[r]), (M << 10) | 0x1f));
#pragma unroll
  for (int r = 0; r < 8; ++r)
    qv[r] += __int_as_float(
        __builtin_amdgcn_ds_swizzle(__float_as_int(qv[r]), (M << 10) | 0x1f));
}

__device__ __forceinline__ void reduce16_butterfly(float (&sv)[8], float (&qv)[8]) {
  swz_step<1>(sv, qv);
  swz_step<2>(sv, qv);
  swz_step<4>(sv, qv);
  swz_step<8>(sv, qv);
}

// ============================================================================
// agg = (1 + eps) * feat     (full overwrite => deterministic across replays)
// ============================================================================
__global__ __launch_bounds__(256) void k_init_agg(float* __restrict__ agg,
                                                  const float* __restrict__ feat,
                                                  const float* __restrict__ eps,
                                                  int n) {
  int i = blockIdx.x * 256 + threadIdx.x;
  float s = 1.0f + eps[0];
  if (i < n) agg[i] = s * feat[i];
}

// ============================================================================
// agg[dst] += feat[src]   — one wave32 per edge, 4 floats per lane.
// Gathers are 128B coalesced bursts; atomics resolve in L2 (the 25.6MB
// feature matrix fits in the 192MB L2), the bandwidth-optimal scatter path.
// ============================================================================
__global__ __launch_bounds__(256) void k_scatter(float* __restrict__ agg,
                                                 const float* __restrict__ feat,
                                                 const long long* __restrict__ ei,
                                                 int E) {
  long long tid = (long long)blockIdx.x * 256 + threadIdx.x;
  int e = (int)(tid >> 5);
  if (e >= E) return;
  int lane = (int)(tid & 31);
  long long s = ei[e];                 // edge_index[0][e]
  long long d = ei[(long long)E + e];  // edge_index[1][e]
  const float* srow = feat + s * 128 + lane * 4;
  float* drow = agg + d * 128 + lane * 4;
  v4f v = *(const v4f*)srow;
  atomicAdd(drow + 0, v.x);
  atomicAdd(drow + 1, v.y);
  atomicAdd(drow + 2, v.z);
  atomicAdd(drow + 3, v.w);
}

// ============================================================================
// Weight prep: f32 row-major (K x Nc) -> bf16 N-major / K-contiguous, so each
// WMMA lane's B-fragment load is two contiguous 16B loads.
// ============================================================================
__global__ __launch_bounds__(256) void k_prep_wt(__bf16* __restrict__ dst,
                                                 const float* __restrict__ src,
                                                 int K, int Nc) {
  int i = blockIdx.x * 256 + threadIdx.x;
  if (i < K * Nc) {
    int k = i / Nc, n = i % Nc;
    dst[n * K + k] = (__bf16)src[i];
  }
}

// ============================================================================
// Fully fused GIN MLP:
//   h = relu(LN( relu(LN(A@W1 + b1; gi, bei)) @ W2 + b2; go, beo ))
// Block: 256 thr = 8 waves; 16 rows x 128 cols.  The inner GEMM's output tile
// is passed to the second GEMM through LDS as bf16 (no global round-trip).
// f32->bf16 conversion of A is folded into LDS staging.
// 4x v_wmma_f32_16x16x32_bf16 per GEMM (K=128).
// C/D layout (ISA 7.12.2): lane%16 = N, VGPR r = row (lanes>=16: rows 8..15).
// ============================================================================
__global__ __launch_bounds__(256) void k_gin_mlp(
    float* __restrict__ out, const float* __restrict__ Af,
    const __bf16* __restrict__ Wt1, const float* __restrict__ b1,
    const float* __restrict__ gi, const float* __restrict__ bei,
    const __bf16* __restrict__ Wt2, const float* __restrict__ b2,
    const float* __restrict__ go, const float* __restrict__ beo, int N) {
  constexpr int STR = 136;
  __shared__ __bf16 lA[16 * STR];
  __shared__ __bf16 lB[16 * STR];
  __shared__ float redS[8][16];
  __shared__ float redQ[8][16];
  __shared__ float mrs[2][16];

  const int t = threadIdx.x;
  const long long row0 = (long long)blockIdx.x * 16;

  {  // stage 16x128 f32 tile, convert to bf16: each thread 8 floats
    int r = t >> 4;
    int cc = (t & 15) * 8;
    v4f f0 = {}, f1 = {};
    if (row0 + r < N) {
      const float* src = Af + (row0 + r) * 128 + cc;
      f0 = *(const v4f*)(src);
      f1 = *(const v4f*)(src + 4);
    }
    v8bf o = {(__bf16)f0.x, (__bf16)f0.y, (__bf16)f0.z, (__bf16)f0.w,
              (__bf16)f1.x, (__bf16)f1.y, (__bf16)f1.z, (__bf16)f1.w};
    *(v8bf*)&lA[r * STR + cc] = o;
  }
  __syncthreads();

  const int wave = t >> 5;
  const int lane = t & 31;
  const int rl = lane & 15;   // row (A) / col (B,C,D) within tile
  const int hs = lane >> 4;   // lane-half select
  const int col = wave * 16 + rl;

  // ---------------- GEMM 1: lA @ W1 ----------------
  v8f c = {};
  {
    const __bf16* wcol = Wt1 + (size_t)col * 128;
#pragma unroll
    for (int k0 = 0; k0 < 128; k0 += 32) {
      BFrag a, b;
      const int ab = k0 + hs * 8;   // A: lanes0-15 K{0..7,16..23}, hi half +8
      a.h[0] = *(const v8bf*)&lA[rl * STR + ab];
      a.h[1] = *(const v8bf*)&lA[rl * STR + ab + 16];
      const int kb = k0 + hs * 8;   // B: lanes0-15 K{0..7,16..23}, hi half +8
      b.h[0] = *(const v8bf*)(wcol + kb);
      b.h[1] = *(const v8bf*)(wcol + kb + 16);
      c = __builtin_amdgcn_wmma_f32_16x16x32_bf16(false, a.v, false, b.v,
                                                  (short)0, c, false, false);
    }
  }

  // ---------------- epilogue 1: bias + LN + ReLU -> bf16 LDS ----------------
  {
    const float bv = b1[col];
    float sv[8], qv[8];
#pragma unroll
    for (int r = 0; r < 8; ++r) {
      c[r] += bv;
      sv[r] = c[r];
      qv[r] = c[r] * c[r];
    }
    reduce16_butterfly(sv, qv);
    if (rl == 0) {
#pragma unroll
      for (int r = 0; r < 8; ++r) {
        redS[wave][hs * 8 + r] = sv[r];
        redQ[wave][hs * 8 + r] = qv[r];
      }
    }
    __syncthreads();
    if (t < 16) {
      float s = 0.f, q = 0.f;
#pragma unroll
      for (int w = 0; w < 8; ++w) { s += redS[w][t]; q += redQ[w][t]; }
      float mean = s * (1.0f / 128.0f);
      float var = q * (1.0f / 128.0f) - mean * mean;
      mrs[0][t] = mean;
      mrs[1][t] = rsqrtf(var + LN_EPS);
    }
    __syncthreads();
    const float gv = gi[col], bev = bei[col];
#pragma unroll
    for (int r = 0; r < 8; ++r) {
      int m = hs * 8 + r;
      float y = (c[r] - mrs[0][m]) * mrs[1][m] * gv + bev;
      lB[m * STR + col] = (__bf16)fmaxf(y, 0.0f);
    }
  }
  __syncthreads();

  // ---------------- GEMM 2: lB @ W2 ----------------
  c = v8f{};
  {
    const __bf16* wcol = Wt2 + (size_t)col * 128;
#pragma unroll
    for (int k0 = 0; k0 < 128; k0 += 32) {
      BFrag a, b;
      const int ab = k0 + hs * 8;
      a.h[0] = *(const v8bf*)&lB[rl * STR + ab];
      a.h[1] = *(const v8bf*)&lB[rl * STR + ab + 16];
      const int kb = k0 + hs * 8;
      b.h[0] = *(const v8bf*)(wcol + kb);
      b.h[1] = *(const v8bf*)(wcol + kb + 16);
      c = __builtin_amdgcn_wmma_f32_16x16x32_bf16(false, a.v, false, b.v,
                                                  (short)0, c, false, false);
    }
  }

  // ---------------- epilogue 2: bias + LN + ReLU -> f32 global --------------
  {
    const float bv = b2[col];
    float sv[8], qv[8];
#pragma unroll
    for (int r = 0; r < 8; ++r) {
      c[r] += bv;
      sv[r] = c[r];
      qv[r] = c[r] * c[r];
    }
    reduce16_butterfly(sv, qv);
    __syncthreads();  // red arrays reused — all waves past epilogue-1 reads
    if (rl == 0) {
#pragma unroll
      for (int r = 0; r < 8; ++r) {
        redS[wave][hs * 8 + r] = sv[r];
        redQ[wave][hs * 8 + r] = qv[r];
      }
    }
    __syncthreads();
    if (t < 16) {
      float s = 0.f, q = 0.f;
#pragma unroll
      for (int w = 0; w < 8; ++w) { s += redS[w][t]; q += redQ[w][t]; }
      float mean = s * (1.0f / 128.0f);
      float var = q * (1.0f / 128.0f) - mean * mean;
      mrs[0][t] = mean;
      mrs[1][t] = rsqrtf(var + LN_EPS);
    }
    __syncthreads();
    const float gv = go[col], bev = beo[col];
#pragma unroll
    for (int r = 0; r < 8; ++r) {
      int m = hs * 8 + r;
      if (row0 + m < N) {
        float y = (c[r] - mrs[0][m]) * mrs[1][m] * gv + bev;
        out[(row0 + m) * 128 + col] = fmaxf(y, 0.0f);
      }
    }
  }
}

// ============================================================================
// Readout: h3 = relu(LN(h2 @ w1 + b1)) [128->32 via WMMA], out = h3 @ w2 + b2
// Block: 64 thr = 2 waves, 16 rows.  f32->bf16 folded into staging.
// ============================================================================
__global__ __launch_bounds__(64) void k_readout(
    float* __restrict__ out, const float* __restrict__ Af,
    const __bf16* __restrict__ Wt /*32 x K128*/, const float* __restrict__ b1,
    const float* __restrict__ g, const float* __restrict__ be,
    const float* __restrict__ w2 /*32x8*/, const float* __restrict__ b2, int N) {
  constexpr int STR = 136;
  __shared__ __bf16 lA[16 * STR];
  __shared__ float redS[2][16];
  __shared__ float redQ[2][16];
  __shared__ float mrs[2][16];
  __shared__ float h3[16][36];

  const int t = threadIdx.x;
  const long long row0 = (long long)blockIdx.x * 16;

  for (int i = t; i < 256; i += 64) {  // 256 chunks of 8 f32 -> 8 bf16
    int r = i >> 4;
    int cc = (i & 15) * 8;
    v4f f0 = {}, f1 = {};
    if (row0 + r < N) {
      const float* src = Af + (row0 + r) * 128 + cc;
      f0 = *(const v4f*)(src);
      f1 = *(const v4f*)(src + 4);
    }
    v8bf o = {(__bf16)f0.x, (__bf16)f0.y, (__bf16)f0.z, (__bf16)f0.w,
              (__bf16)f1.x, (__bf16)f1.y, (__bf16)f1.z, (__bf16)f1.w};
    *(v8bf*)&lA[r * STR + cc] = o;
  }
  __syncthreads();

  const int wave = t >> 5;
  const int lane = t & 31;
  const int rl = lane & 15;
  const int hs = lane >> 4;
  const int col = wave * 16 + rl;

  v8f c = {};
  const __bf16* wcol = Wt + (size_t)col * 128;
#pragma unroll
  for (int k0 = 0; k0 < 128; k0 += 32) {
    BFrag a, b;
    const int ab = k0 + hs * 8;
    a.h[0] = *(const v8bf*)&lA[rl * STR + ab];
    a.h[1] = *(const v8bf*)&lA[rl * STR + ab + 16];
    const int kb = k0 + hs * 8;
    b.h[0] = *(const v8bf*)(wcol + kb);
    b.h[1] = *(const v8bf*)(wcol + kb + 16);
    c = __builtin_amdgcn_wmma_f32_16x16x32_bf16(false, a.v, false, b.v,
                                                (short)0, c, false, false);
  }

  const float bv = b1[col];
  float sv[8], qv[8];
#pragma unroll
  for (int r = 0; r < 8; ++r) {
    c[r] += bv;
    sv[r] = c[r];
    qv[r] = c[r] * c[r];
  }
  reduce16_butterfly(sv, qv);
  if (rl == 0) {
#pragma unroll
    for (int r = 0; r < 8; ++r) {
      redS[wave][hs * 8 + r] = sv[r];
      redQ[wave][hs * 8 + r] = qv[r];
    }
  }
  __syncthreads();

  if (t < 16) {
    float s = redS[0][t] + redS[1][t];
    float q = redQ[0][t] + redQ[1][t];
    float mean = s * (1.0f / 32.0f);
    float var = q * (1.0f / 32.0f) - mean * mean;
    mrs[0][t] = mean;
    mrs[1][t] = rsqrtf(var + LN_EPS);
  }
  __syncthreads();

  const float gv = g[col], bev = be[col];
#pragma unroll
  for (int r = 0; r < 8; ++r) {
    int m = hs * 8 + r;
    float y = (c[r] - mrs[0][m]) * mrs[1][m] * gv + bev;
    h3[m][col] = fmaxf(y, 0.0f);
  }
  __syncthreads();

  // final tiny GEMM: 16 rows x 8 outputs, K=32
  for (int i = t; i < 128; i += 64) {
    int r = i >> 3, j = i & 7;
    if (row0 + r < N) {
      float acc = b2[j];
#pragma unroll
      for (int k = 0; k < 32; ++k) acc += h3[r][k] * w2[k * 8 + j];
      out[(row0 + r) * 8 + j] = acc;
    }
  }
}

// ============================================================================
// host launcher
// ============================================================================
extern "C" void kernel_launch(void* const* d_in, const int* in_sizes, int n_in,
                              void* d_out, int out_size, void* d_ws, size_t ws_size,
                              hipStream_t stream) {
  (void)n_in; (void)out_size; (void)ws_size;
  const float* x       = (const float*)d_in[0];
  const long long* ei  = (const long long*)d_in[1];
  const int E = in_sizes[2];            // edge_weight count == E
  const int N = in_sizes[0] / 128;

  const float* eps1 = (const float*)d_in[3];
  const float* m1w1 = (const float*)d_in[4];
  const float* m1b1 = (const float*)d_in[5];
  const float* m1g  = (const float*)d_in[6];
  const float* m1be = (const float*)d_in[7];
  const float* m1w2 = (const float*)d_in[8];
  const float* m1b2 = (const float*)d_in[9];
  const float* g1   = (const float*)d_in[10];
  const float* be1  = (const float*)d_in[11];
  const float* eps2 = (const float*)d_in[12];
  const float* m2w1 = (const float*)d_in[13];
  const float* m2b1 = (const float*)d_in[14];
  const float* m2g  = (const float*)d_in[15];
  const float* m2be = (const float*)d_in[16];
  const float* m2w2 = (const float*)d_in[17];
  const float* m2b2 = (const float*)d_in[18];
  const float* g2   = (const float*)d_in[19];
  const float* be2  = (const float*)d_in[20];
  const float* row1 = (const float*)d_in[21];
  const float* rob1 = (const float*)d_in[22];
  const float* rog  = (const float*)d_in[23];
  const float* robe = (const float*)d_in[24];
  const float* row2 = (const float*)d_in[25];
  const float* rob2 = (const float*)d_in[26];

  // workspace carve (~51.3 MB)
  char* p = (char*)d_ws;
  float* agg = (float*)p;  p += (size_t)N * 128 * sizeof(float);
  float* h   = (float*)p;  p += (size_t)N * 128 * sizeof(float);
  __bf16* wt1 = (__bf16*)p; p += 128 * 128 * sizeof(__bf16);
  __bf16* wt2 = (__bf16*)p; p += 128 * 128 * sizeof(__bf16);
  __bf16* rwt = (__bf16*)p; p += 128 * 32 * sizeof(__bf16);

  const int elems = N * 128;
  const int gElem = (elems + 255) / 256;
  const long long sthreads = (long long)E * 32;
  const int gScat = (int)((sthreads + 255) / 256);
  const int gGemm = (N + 15) / 16;
  const int gW128 = (128 * 128 + 255) / 256;
  const int gW32  = (128 * 32 + 255) / 256;

  // ---------------- Layer 1 ----------------
  k_prep_wt<<<gW128, 256, 0, stream>>>(wt1, m1w1, 128, 128);
  k_prep_wt<<<gW128, 256, 0, stream>>>(wt2, m1w2, 128, 128);
  k_init_agg<<<gElem, 256, 0, stream>>>(agg, x, eps1, elems);
  k_scatter<<<gScat, 256, 0, stream>>>(agg, x, ei, E);
  k_gin_mlp<<<gGemm, 256, 0, stream>>>(h, agg, wt1, m1b1, m1g, m1be,
                                       wt2, m1b2, g1, be1, N);

  // ---------------- Layer 2 ----------------
  k_prep_wt<<<gW128, 256, 0, stream>>>(wt1, m2w1, 128, 128);
  k_prep_wt<<<gW128, 256, 0, stream>>>(wt2, m2w2, 128, 128);
  k_init_agg<<<gElem, 256, 0, stream>>>(agg, h, eps2, elems);
  k_scatter<<<gScat, 256, 0, stream>>>(agg, h, ei, E);
  k_gin_mlp<<<gGemm, 256, 0, stream>>>(h, agg, wt1, m2b1, m2g, m2be,
                                       wt2, m2b2, g2, be2, N);

  // ---------------- Readout ----------------
  k_prep_wt<<<gW32, 256, 0, stream>>>(rwt, row1, 128, 32);
  k_readout<<<gGemm, 64, 0, stream>>>((float*)d_out, h, rwt, rob1, rog, robe,
                                      row2, rob2, N);
}
